// TwoFWLConv_68436008895100
// MI455X (gfx1250) — matrix operand order repaired and checked by reading, not compiled, not run on hardware
//
#include <hip/hip_runtime.h>

// ---------------------------------------------------------------------------
// TwoFWLConv for MI455X (gfx1250): bf16 WMMA pipeline + TDM (tensor DMA) staging.
//   h1 = mask * relu(X @ W1 + b1)   (bf16, ws)
//   h2 = mask * relu(X @ W2 + b2)   (bf16, ws)
//   out[b,i,j,d] = mask * sum_k h1[b,i,k,d] * h2[b,k,j,d]   (f32)
// B=128, N=32, D=256.
// ---------------------------------------------------------------------------

typedef __bf16 v16bf __attribute__((ext_vector_type(16)));
typedef float  v8f   __attribute__((ext_vector_type(8)));
typedef __bf16 bf4   __attribute__((ext_vector_type(4)));
typedef __bf16 bf8   __attribute__((ext_vector_type(8)));
typedef float  f32x4 __attribute__((ext_vector_type(4)));
typedef unsigned int u32x4 __attribute__((ext_vector_type(4)));
typedef int    i32x8 __attribute__((ext_vector_type(8)));
typedef int    i32x4 __attribute__((ext_vector_type(4)));

#define DD   256      // embedding dim
#define NN   32       // nodes per graph
#define ROWS 131072   // B*N*N rows of the per-tuple GEMM

static __device__ __forceinline__ v8f wmma_bf16(v16bf a, v16bf b, v8f c) {
  return __builtin_amdgcn_wmma_f32_16x16x32_bf16(
      /*neg_a=*/false, a, /*neg_b=*/false, b,
      /*c_mod=*/(short)0, c, /*reuse_a=*/false, /*reuse_b=*/false);
}

// ---------------------------------------------------------------------------
// Issue one TDM 2D tile load: global (bf16 tensor, row stride stride0 elems)
// -> LDS at byte offset lds_off. Tile: tile_d0 elems (contig) x tile_d1 rows.
// Descriptor layout per CDNA5 ISA 8.3-8.6; groups 2/3 zero (2D tensor).
// Toolchain uses the 6-arg builtin: (u32x4, i32x8, i32x4, i32x4, i32x8, cpol).
// ---------------------------------------------------------------------------
static __device__ __forceinline__ void tdm_load_2d(
    unsigned lds_off, const void* gptr,
    unsigned tensor_d0, unsigned tensor_d1,
    unsigned tile_d0, unsigned tile_d1, unsigned stride0) {
  unsigned long long ga = (unsigned long long)gptr;
  u32x4 g0;
  g0[0] = 1u;                                           // count=1 (valid), user mode
  g0[1] = lds_off;                                      // lds_addr[31:0]
  g0[2] = (unsigned)(ga & 0xffffffffu);                 // global_addr[31:0]
  g0[3] = (unsigned)((ga >> 32) & 0x01ffffffu)          // global_addr[56:32]
          | (2u << 30);                                 // type=2 ("image")
  i32x8 g1;
  g1[0] = (int)(1u << 16);                              // data_size=1 -> 2 bytes
  g1[1] = (int)((tensor_d0 & 0xffffu) << 16);           // tensor_dim0[15:0]
  g1[2] = (int)((tensor_d0 >> 16) |
                ((tensor_d1 & 0xffffu) << 16));         // dim0[31:16] | dim1[15:0]
  g1[3] = (int)((tensor_d1 >> 16) | (tile_d0 << 16));   // dim1[31:16] | tile_dim0
  g1[4] = (int)(tile_d1 & 0xffffu);                     // tile_dim1 | tile_dim2=0
  g1[5] = (int)stride0;                                 // tensor_dim0_stride[31:0]
  g1[6] = 0;                                            // stride0[47:32] | stride1 lo
  g1[7] = 0;                                            // stride1 hi
  i32x4 z4 = {0, 0, 0, 0};
  i32x8 z8 = {0, 0, 0, 0, 0, 0, 0, 0};
  __builtin_amdgcn_tensor_load_to_lds(g0, g1, z4, z4, z8, 0);
}

// --------------------------------------------------------------------------
// Kernel 0: W1/W2 (f32, [c][d] row-major) -> bf16 transposed [d][c] in ws.
// --------------------------------------------------------------------------
__global__ __launch_bounds__(256) void prep_weights(
    const float* __restrict__ W1, const float* __restrict__ W2,
    __bf16* __restrict__ W1t, __bf16* __restrict__ W2t) {
  int idx = blockIdx.x * 256 + threadIdx.x;     // 0 .. 65535
  int c = idx >> 8;
  int d = idx & 255;
  W1t[d * DD + c] = (__bf16)W1[c * DD + d];
  W2t[d * DD + c] = (__bf16)W2[c * DD + d];
}

// --------------------------------------------------------------------------
// Kernel 1: fused per-tuple linear. Block tile: 64 rows x 256 cols.
// 8 waves arranged 2(m) x 4(n): each wave owns 32x64 (2x4 WMMA tiles).
// blockIdx.x : row block (0..2047), blockIdx.y : 0 -> (W1,b1,h1), 1 -> (W2,b2,h2)
// LDS: ldsA 64x256 bf16 (32KB, VALU-converted) + ldsW 256x256 bf16 (128KB, TDM).
// --------------------------------------------------------------------------
__global__ __launch_bounds__(256) void fused_lin(
    const float* __restrict__ X,
    const float* __restrict__ bias1, const float* __restrict__ bias2,
    const int* __restrict__ mask,
    const __bf16* __restrict__ W1t, const __bf16* __restrict__ W2t,
    __bf16* __restrict__ h1, __bf16* __restrict__ h2) {
  extern __shared__ char smem[];
  __bf16* ldsA = (__bf16*)smem;                 // [64][256]
  __bf16* ldsW = (__bf16*)(smem + 64 * DD * 2); // [256 n][256 k]

  const int t    = threadIdx.x;
  const int lane = t & 31;
  const int wid  = t >> 5;
  const int row0 = blockIdx.x * 64;
  const int sel  = blockIdx.y;

  const float*  bias = sel ? bias2 : bias1;
  const __bf16* Wt   = sel ? W2t : W1t;
  __bf16*       h    = sel ? h2 : h1;

  // --- kick off async TDM load of the full bf16 weight tile into ldsW
  if (wid == 0) {
    unsigned wofs = (unsigned)(unsigned long long)(uintptr_t)ldsW;
    tdm_load_2d(wofs, Wt, DD, DD, DD, DD, DD);
  }

  // --- meanwhile: stage A tile (64 contiguous rows of X), f32 -> bf16 on VALU
  {
    const f32x4* src = (const f32x4*)(X + (size_t)row0 * DD);
    for (int i = t; i < 64 * DD / 4; i += 256) {
      f32x4 v = src[i];
      bf4 b;
      b[0] = (__bf16)v[0]; b[1] = (__bf16)v[1];
      b[2] = (__bf16)v[2]; b[3] = (__bf16)v[3];
      *(bf4*)(ldsA + i * 4) = b;
    }
  }
  if (wid == 0) __builtin_amdgcn_s_wait_tensorcnt(0);
  __syncthreads();

  const int wm    = wid >> 2;       // 0..1
  const int wn    = wid & 3;        // 0..3
  const int mbase = wm * 32;
  const int nbase = wn * 64;
  const int hi    = lane >> 4;      // half-wave
  const int l15   = lane & 15;

  v8f acc[2][4];
#pragma unroll
  for (int tm = 0; tm < 2; ++tm)
#pragma unroll
    for (int tn = 0; tn < 4; ++tn) {
      v8f z = {0.f, 0.f, 0.f, 0.f, 0.f, 0.f, 0.f, 0.f};
      acc[tm][tn] = z;
    }

  for (int ks = 0; ks < DD; ks += 32) {
    // A fragments (16-bit A layout: half-wave hi=0 -> K 0..7,16..23; hi=1 -> +8)
    v16bf afrag[2];
#pragma unroll
    for (int tm = 0; tm < 2; ++tm) {
      const __bf16* ap = ldsA + (size_t)(mbase + tm * 16 + l15) * DD + ks + hi * 8;
      bf8 lo = *(const bf8*)ap;           // K = ks + hi*8 + 0..7
      bf8 hh = *(const bf8*)(ap + 16);    // K = ks + 16 + hi*8 + 0..7
      v16bf a;
#pragma unroll
      for (int e = 0; e < 8; ++e) { a[e] = lo[e]; a[e + 8] = hh[e]; }
      afrag[tm] = a;
    }
    // B fragments (same K order as A: hi=0 -> K 0..7,16..23; hi=1 -> +8)
    v16bf bfrag[4];
#pragma unroll
    for (int tn = 0; tn < 4; ++tn) {
      const __bf16* bp = ldsW + (size_t)(nbase + tn * 16 + l15) * DD + ks + hi * 8;
      bf8 lo = *(const bf8*)bp;
      bf8 hh = *(const bf8*)(bp + 16);
      v16bf b;
#pragma unroll
      for (int e = 0; e < 8; ++e) { b[e] = lo[e]; b[e + 8] = hh[e]; }
      bfrag[tn] = b;
    }
#pragma unroll
    for (int tm = 0; tm < 2; ++tm)
#pragma unroll
      for (int tn = 0; tn < 4; ++tn)
        acc[tm][tn] = wmma_bf16(afrag[tm], bfrag[tn], acc[tm][tn]);
  }

  // --- epilogue: bias + relu + tuple mask, store bf16
#pragma unroll
  for (int tm = 0; tm < 2; ++tm) {
#pragma unroll
    for (int tn = 0; tn < 4; ++tn) {
      v8f c = acc[tm][tn];
#pragma unroll
      for (int r = 0; r < 8; ++r) {
        int M = mbase + tm * 16 + r + hi * 8;
        int N = nbase + tn * 16 + l15;
        int grow = row0 + M;                  // flat (b,i,k) tuple index
        float v = c[r] + bias[N];
        v = v > 0.f ? v : 0.f;
        v = mask[grow] ? v : 0.f;
        h[(size_t)grow * DD + N] = (__bf16)v;
      }
    }
  }
}

// --------------------------------------------------------------------------
// Kernel 2: 2-FWL contraction. One block = (graph b, 32-wide d chunk).
// LDS slabs staged by TDM: L1[i][k][dd], L2[k][j][dd] (dd innermost, 32 wide),
// each a 2D tile (32 elems x 1024 rows, row stride 256) of h1[b]/h2[b].
// Each of 8 waves owns 4 d-planes: K=32 == one bf16 WMMA; 2x2 output tiles.
// One 8B LDS read feeds the same fragment element of 4 d-planes.
// --------------------------------------------------------------------------
__global__ __launch_bounds__(256) void contract(
    const __bf16* __restrict__ h1, const __bf16* __restrict__ h2,
    const int* __restrict__ mask, float* __restrict__ out) {
  extern __shared__ char smem[];
  __bf16* L1 = (__bf16*)smem;                     // [1024][32]
  __bf16* L2 = (__bf16*)(smem + 1024 * 32 * 2);

  const int t    = threadIdx.x;
  const int lane = t & 31;
  const int wid  = t >> 5;
  const int b    = blockIdx.x >> 3;     // 0..127
  const int d0   = (blockIdx.x & 7) * 32;

  // --- TDM: DMA both slabs into LDS (wave 0 issues, TENSORcnt tracked)
  const __bf16* g1p = h1 + (size_t)b * 1024 * DD + d0;
  const __bf16* g2p = h2 + (size_t)b * 1024 * DD + d0;
  if (wid == 0) {
    unsigned l1o = (unsigned)(unsigned long long)(uintptr_t)L1;
    unsigned l2o = (unsigned)(unsigned long long)(uintptr_t)L2;
    tdm_load_2d(l1o, g1p, DD, 1024, 32, 1024, DD);
    tdm_load_2d(l2o, g2p, DD, 1024, 32, 1024, DD);
    __builtin_amdgcn_s_wait_tensorcnt(0);
  }
  __syncthreads();

  const int dd0 = wid * 4;          // this wave's 4 d-planes within the chunk
  const int hi  = lane >> 4;
  const int l15 = lane & 15;

  v16bf A[2][4];   // [i-tile][d-plane]
  v16bf Bf[2][4];  // [j-tile][d-plane]
  v8f   acc[4][2][2];
#pragma unroll
  for (int dd = 0; dd < 4; ++dd)
#pragma unroll
    for (int ti = 0; ti < 2; ++ti)
#pragma unroll
      for (int tj = 0; tj < 2; ++tj) {
        v8f z = {0.f, 0.f, 0.f, 0.f, 0.f, 0.f, 0.f, 0.f};
        acc[dd][ti][tj] = z;
      }

  // --- gather fragments: element e of each frag for 4 d-planes per 8B read
#pragma unroll
  for (int e = 0; e < 16; ++e) {
    const int kA = e + (e & 8) + hi * 8;   // A 16-bit layout K index
    const int kB = e + (e & 8) + hi * 8;   // B K index (same order as A)
#pragma unroll
    for (int ti = 0; ti < 2; ++ti) {
      int m = ti * 16 + l15;               // i row
      bf4 va = *(const bf4*)(L1 + ((size_t)(m * NN + kA)) * 32 + dd0);
#pragma unroll
      for (int dd = 0; dd < 4; ++dd) A[ti][dd][e] = va[dd];
    }
#pragma unroll
    for (int tj = 0; tj < 2; ++tj) {
      int n = tj * 16 + l15;               // j col
      bf4 vb = *(const bf4*)(L2 + ((size_t)(kB * NN + n)) * 32 + dd0);
#pragma unroll
      for (int dd = 0; dd < 4; ++dd) Bf[tj][dd][e] = vb[dd];
    }
  }

  // --- 16 WMMAs: full K=32 contraction per (d-plane, output tile)
#pragma unroll
  for (int dd = 0; dd < 4; ++dd)
#pragma unroll
    for (int ti = 0; ti < 2; ++ti)
#pragma unroll
      for (int tj = 0; tj < 2; ++tj)
        acc[dd][ti][tj] = wmma_bf16(A[ti][dd], Bf[tj][dd], acc[dd][ti][tj]);

  // --- masked float4 stores (4 consecutive d per store)
  float* ob = out + (size_t)b * 1024 * DD + d0 + dd0;
  const int* mrow = mask + b * 1024;
#pragma unroll
  for (int ti = 0; ti < 2; ++ti) {
#pragma unroll
    for (int tj = 0; tj < 2; ++tj) {
#pragma unroll
      for (int r = 0; r < 8; ++r) {
        int i = ti * 16 + r + hi * 8;
        int j = tj * 16 + l15;
        f32x4 v;
        v[0] = acc[0][ti][tj][r];
        v[1] = acc[1][ti][tj][r];
        v[2] = acc[2][ti][tj][r];
        v[3] = acc[3][ti][tj][r];
        if (!mrow[i * NN + j]) { v[0] = 0.f; v[1] = 0.f; v[2] = 0.f; v[3] = 0.f; }
        *(f32x4*)(ob + (size_t)(i * NN + j) * DD) = v;
      }
    }
  }
}

// --------------------------------------------------------------------------
extern "C" void kernel_launch(void* const* d_in, const int* in_sizes, int n_in,
                              void* d_out, int out_size, void* d_ws, size_t ws_size,
                              hipStream_t stream) {
  const float* X    = (const float*)d_in[0];
  const float* W1   = (const float*)d_in[1];
  const float* b1   = (const float*)d_in[2];
  const float* W2   = (const float*)d_in[3];
  const float* b2   = (const float*)d_in[4];
  const int*   mask = (const int*)d_in[5];
  float* out = (float*)d_out;

  char* ws = (char*)d_ws;
  __bf16* W1t = (__bf16*)ws;                               // 128 KB
  __bf16* W2t = (__bf16*)(ws + 131072);                    // 128 KB
  __bf16* h1  = (__bf16*)(ws + 262144);                    // 64 MB
  __bf16* h2  = (__bf16*)(ws + 262144 + (size_t)ROWS * DD * 2);

  prep_weights<<<256, 256, 0, stream>>>(W1, W2, W1t, W2t);

  dim3 g1(ROWS / 64, 2);
  fused_lin<<<g1, 256, (64 * DD + DD * DD) * 2, stream>>>(
      X, b1, b2, mask, W1t, W2t, h1, h2);

  contract<<<128 * 8, 256, 2 * 1024 * 32 * 2, stream>>>(h1, h2, mask, out);
}
